// GIN0_8632884265157
// MI455X (gfx1250) — compile-verified, matrix operand order checked by reading, not run on hardware
//
#include <hip/hip_runtime.h>
#include <math.h>

// ---------------- problem constants ----------------
#define N_NODES   100000
#define N_EDGES   1600000
#define N_GRAPHS  512
#define FEAT      128            // N_FEAT == HIDDEN == 128
#define N_LAYERS  4
#define N_CLASSES 6
#define BN_EPS    1e-5f

#define MTILES    (N_NODES / 16)   // 6250 (exact)
#define KCHUNKS   (FEAT / 32)      // 4
#define NTILES    (FEAT / 16)      // 8
#define FRAG_HALF 16               // halves per lane per fragment
#define FRAG_SZ   (32 * FRAG_HALF) // halves per 16x32 fragment (one wave)
#define WMAT_SZ   (NTILES * KCHUNKS * FRAG_SZ) // 16384 halves per packed 128x128 weight

typedef __attribute__((ext_vector_type(16))) _Float16 v16h;
typedef __attribute__((ext_vector_type(8)))  float    v8f;
typedef int v4i __attribute__((vector_size(16)));      // matches async builtin param type

#define AS_LDS  __attribute__((address_space(3)))
#define AS_GLB  __attribute__((address_space(1)))

// ---------------- workspace layout (bytes) ----------------
#define WS_Z      0
#define WS_Y      51200000ULL
#define WS_H      102400000ULL
#define WS_APACK  153600000ULL
#define WS_WPACK  179200000ULL
#define WS_STATS  179462144ULL
#define WS_G      179463168ULL
#define WS_ACT1   180511744ULL

// K-offset within a 32-wide K chunk for fragment element j of a lane
// (ISA 7.12.2, 16-bit A 16x32: lanes 0-15 hold K {0..7,16..23}, lanes 16-31 hold K {8..15,24..31})
__device__ __forceinline__ int frag_k(int lane, int j) {
    return ((j < 8) ? j : j + 8) + ((lane >= 16) ? 8 : 0);
}

// 16-byte global -> LDS copy; async (GLOBAL_LOAD_ASYNC_TO_LDS_B128, ASYNCcnt) when the
// toolchain exposes the gfx1250 builtin, plain LDS store otherwise.
__device__ __forceinline__ void stage16(const _Float16* __restrict__ gsrc, _Float16* ldst) {
#if __has_builtin(__builtin_amdgcn_global_load_async_to_lds_b128)
    __builtin_amdgcn_global_load_async_to_lds_b128(
        (AS_GLB v4i*)gsrc, (AS_LDS v4i*)ldst, 0, 0);
#else
    *(float4*)ldst = *(const float4*)gsrc;
#endif
}

__device__ __forceinline__ void stage_wait() {
#if __has_builtin(__builtin_amdgcn_s_wait_asynccnt)
    __builtin_amdgcn_s_wait_asynccnt(0);
#endif
    __syncthreads();
}

// ---------------- weight packing (fp32 row-major [K=128][N=128] -> B fragments) ----------------
__global__ void pack_w_kernel(const float* __restrict__ W, _Float16* __restrict__ Wp) {
    int t = blockIdx.x * blockDim.x + threadIdx.x;  // 0..16383
    if (t >= WMAT_SZ) return;
    int j    = t & 15;
    int lane = (t >> 4) & 31;
    int kc   = (t >> 9) & 3;
    int nt   = (t >> 11) & 7;
    int n = nt * 16 + (lane & 15);
    int k = kc * 32 + frag_k(lane, j);
    Wp[t] = (_Float16)W[k * FEAT + n];
}

// ---------------- z = h (float4 copy) ----------------
__global__ void copy_f4_kernel(const float4* __restrict__ src, float4* __restrict__ dst, int n4) {
    int t = blockIdx.x * blockDim.x + threadIdx.x;
    if (t < n4) dst[t] = src[t];
}

// ---------------- edge scatter-add: z[dst] += h[src] ----------------
__global__ void scatter_kernel(const float* __restrict__ h, const int* __restrict__ ei,
                               float* __restrict__ z) {
    int t = blockIdx.x * blockDim.x + threadIdx.x;  // N_EDGES*32 threads
    int e = t >> 5;
    if (e >= N_EDGES) return;
    int c = (t & 31) << 2;
    int s = ei[e];
    int d = ei[N_EDGES + e];
    const float4 v = *(const float4*)(h + (size_t)s * FEAT + c);
    float* zp = z + (size_t)d * FEAT + c;
    atomicAdd(zp + 0, v.x);
    atomicAdd(zp + 1, v.y);
    atomicAdd(zp + 2, v.z);
    atomicAdd(zp + 3, v.w);
}

// ---------------- pack fp32 activations -> f16 A fragments ----------------
__global__ void packA_kernel(const float* __restrict__ z, _Float16* __restrict__ Ap) {
    int t = blockIdx.x * blockDim.x + threadIdx.x;  // MTILES*4*32 threads
    if (t >= MTILES * KCHUNKS * 32) return;
    int lane = t & 31;
    int kc   = (t >> 5) & 3;
    int mt   = t >> 7;
    int m     = mt * 16 + (lane & 15);
    int kbase = kc * 32 + ((lane >= 16) ? 8 : 0);
    const float* row = z + (size_t)m * FEAT;
    _Float16* o = Ap + (size_t)t * FRAG_HALF;
#pragma unroll
    for (int j = 0; j < 8; ++j) o[j]     = (_Float16)row[kbase + j];
#pragma unroll
    for (int j = 0; j < 8; ++j) o[8 + j] = (_Float16)row[kbase + 16 + j];
}

// ---------------- fused BatchNorm + ReLU + pack -> f16 A fragments ----------------
__global__ void packA_bn_kernel(const float* __restrict__ y, const float* __restrict__ stats,
                                const float* __restrict__ gamma, const float* __restrict__ beta,
                                _Float16* __restrict__ Ap) {
    int t = blockIdx.x * blockDim.x + threadIdx.x;
    if (t >= MTILES * KCHUNKS * 32) return;
    int lane = t & 31;
    int kc   = (t >> 5) & 3;
    int mt   = t >> 7;
    int m     = mt * 16 + (lane & 15);
    int kbase = kc * 32 + ((lane >= 16) ? 8 : 0);
    const float* row = y + (size_t)m * FEAT;
    _Float16* o = Ap + (size_t)t * FRAG_HALF;
    const float invN = 1.0f / (float)N_NODES;
#pragma unroll
    for (int jj = 0; jj < 16; ++jj) {
        int k  = kbase + ((jj < 8) ? jj : 8 + jj);  // {0..7} then {16..23} relative
        float mu  = stats[k] * invN;
        float var = stats[FEAT + k] * invN - mu * mu;
        float v = (row[k] - mu) * rsqrtf(var + BN_EPS) * gamma[k] + beta[k];
        o[jj] = (_Float16)(v > 0.0f ? v : 0.0f);
    }
}

// ---------------- WMMA GEMM: y[M,128] = A[M,128] @ W[128,128] + b ----------------
// Weight (32 KB, fragment layout) staged once per workgroup into LDS via async
// global->LDS copies; 8 waves read B fragments from LDS (ds_load), A streamed
// from global, f32 accumulate in v_wmma_f32_16x16x32_f16.
__global__ __launch_bounds__(256)
void gemm_wmma_kernel(const _Float16* __restrict__ Ap, const _Float16* __restrict__ Wp,
                      const float* __restrict__ bias, float* __restrict__ y) {
    __shared__ alignas(32) _Float16 Bs[WMAT_SZ];   // 32 KB

    // cooperative stage: 256 threads x 128 B each (8 x b128)
    int tid = threadIdx.x;
#pragma unroll
    for (int i = 0; i < 8; ++i) {
        int off = tid * 64 + i * 8;               // halves; 16 B per op
        stage16(Wp + off, Bs + off);
    }
    stage_wait();

    int wave = tid >> 5;
    int lane = tid & 31;
    int mt = blockIdx.x * 8 + wave;
    if (mt >= MTILES) return;   // wave-uniform: EXEC stays all-1 for WMMA

    const v16h* A = (const v16h*)(Ap + (size_t)mt * KCHUNKS * FRAG_SZ);
    v16h a0 = A[0 * 32 + lane];
    v16h a1 = A[1 * 32 + lane];
    v16h a2 = A[2 * 32 + lane];
    v16h a3 = A[3 * 32 + lane];

    int col0    = lane & 15;
    int rowbase = mt * 16 + ((lane >= 16) ? 8 : 0);

    for (int nt = 0; nt < NTILES; ++nt) {
        const v16h* B = (const v16h*)(Bs + (size_t)nt * KCHUNKS * FRAG_SZ);
        v8f c = {};
        c = __builtin_amdgcn_wmma_f32_16x16x32_f16(false, a0, false, B[0 * 32 + lane], (short)0, c, false, false);
        c = __builtin_amdgcn_wmma_f32_16x16x32_f16(false, a1, false, B[1 * 32 + lane], (short)0, c, false, false);
        c = __builtin_amdgcn_wmma_f32_16x16x32_f16(false, a2, false, B[2 * 32 + lane], (short)0, c, false, false);
        c = __builtin_amdgcn_wmma_f32_16x16x32_f16(false, a3, false, B[3 * 32 + lane], (short)0, c, false, false);
        int   col = nt * 16 + col0;
        float bv  = bias[col];
#pragma unroll
        for (int r = 0; r < 8; ++r)
            y[(size_t)(rowbase + r) * FEAT + col] = c[r] + bv;
    }
}

// ---------------- BN column statistics: sum / sumsq per feature ----------------
__global__ void bn_stats_kernel(const float* __restrict__ y, float* __restrict__ stats) {
    int c  = threadIdx.x;            // 128 threads/block, one column each
    int r0 = blockIdx.x * 256;
    int r1 = r0 + 256; if (r1 > N_NODES) r1 = N_NODES;
    float s = 0.0f, s2 = 0.0f;
    for (int r = r0; r < r1; ++r) {
        float v = y[(size_t)r * FEAT + c];
        s += v; s2 += v * v;
    }
    atomicAdd(&stats[c], s);
    atomicAdd(&stats[FEAT + c], s2);
}

// ---------------- BN + ReLU -> fp32 h ----------------
__global__ void bn_relu_kernel(const float* __restrict__ y, const float* __restrict__ stats,
                               const float* __restrict__ gamma, const float* __restrict__ beta,
                               float* __restrict__ h) {
    int t = blockIdx.x * blockDim.x + threadIdx.x;
    if (t >= N_NODES * FEAT) return;
    int c = t & (FEAT - 1);
    const float invN = 1.0f / (float)N_NODES;
    float mu  = stats[c] * invN;
    float var = stats[FEAT + c] * invN - mu * mu;
    float v = (y[t] - mu) * rsqrtf(var + BN_EPS) * gamma[c] + beta[c];
    h[t] = v > 0.0f ? v : 0.0f;
}

// ---------------- global_add_pool into g's column block ----------------
__global__ void pool_kernel(const float* __restrict__ h, const int* __restrict__ batch,
                            float* __restrict__ g, int colOff) {
    int t = blockIdx.x * blockDim.x + threadIdx.x;  // N_NODES*32 threads
    int node = t >> 5;
    if (node >= N_NODES) return;
    int c  = (t & 31) << 2;
    int gi = batch[node];
    const float4 v = *(const float4*)(h + (size_t)node * FEAT + c);
    float* gp = g + (size_t)gi * (N_LAYERS * FEAT) + colOff + c;
    atomicAdd(gp + 0, v.x);
    atomicAdd(gp + 1, v.y);
    atomicAdd(gp + 2, v.z);
    atomicAdd(gp + 3, v.w);
}

// ---------------- lin1: act = relu(g @ W + b), g:[512,512], W:[512,128] ----------------
__global__ void lin1_kernel(const float* __restrict__ g, const float* __restrict__ W,
                            const float* __restrict__ b, float* __restrict__ act) {
    int t = blockIdx.x * blockDim.x + threadIdx.x;  // 512*128 threads
    if (t >= N_GRAPHS * FEAT) return;
    int j = t & (FEAT - 1);
    int i = t >> 7;
    float s = b[j];
    const float* gr = g + (size_t)i * (N_LAYERS * FEAT);
    for (int k = 0; k < N_LAYERS * FEAT; ++k)
        s += gr[k] * W[(size_t)k * FEAT + j];
    act[t] = s > 0.0f ? s : 0.0f;
}

// ---------------- lin2 + log_softmax ----------------
__global__ void head_kernel(const float* __restrict__ act, const float* __restrict__ W,
                            const float* __restrict__ b, float* __restrict__ out) {
    int i = blockIdx.x * blockDim.x + threadIdx.x;
    if (i >= N_GRAPHS) return;
    float lg[N_CLASSES];
    const float* ar = act + (size_t)i * FEAT;
#pragma unroll
    for (int c = 0; c < N_CLASSES; ++c) {
        float s = b[c];
        for (int k = 0; k < FEAT; ++k) s += ar[k] * W[k * N_CLASSES + c];
        lg[c] = s;
    }
    float m = lg[0];
#pragma unroll
    for (int c = 1; c < N_CLASSES; ++c) m = lg[c] > m ? lg[c] : m;
    float sum = 0.0f;
#pragma unroll
    for (int c = 0; c < N_CLASSES; ++c) sum += __expf(lg[c] - m);
    float lse = __logf(sum);
#pragma unroll
    for (int c = 0; c < N_CLASSES; ++c) out[(size_t)i * N_CLASSES + c] = lg[c] - m - lse;
}

// =====================================================================

extern "C" void kernel_launch(void* const* d_in, const int* in_sizes, int n_in,
                              void* d_out, int out_size, void* d_ws, size_t ws_size,
                              hipStream_t stream) {
    const float* x     = (const float*)d_in[0];
    const int*   ei    = (const int*)d_in[1];
    const int*   batch = (const int*)d_in[2];
    // layer l params at d_in[3 + l*8 + {0:W1,1:b1,2:g1,3:be1,4:W2,5:b2,6:g2,7:be2}]
    const float* lin1_W = (const float*)d_in[3 + N_LAYERS * 8 + 0];
    const float* lin1_b = (const float*)d_in[3 + N_LAYERS * 8 + 1];
    const float* lin2_W = (const float*)d_in[3 + N_LAYERS * 8 + 2];
    const float* lin2_b = (const float*)d_in[3 + N_LAYERS * 8 + 3];

    char* ws = (char*)d_ws;
    float*    z     = (float*)(ws + WS_Z);
    float*    y     = (float*)(ws + WS_Y);
    float*    h     = (float*)(ws + WS_H);
    _Float16* Apack = (_Float16*)(ws + WS_APACK);
    _Float16* Wpack = (_Float16*)(ws + WS_WPACK);
    float*    stats = (float*)(ws + WS_STATS);
    float*    g     = (float*)(ws + WS_G);
    float*    act1  = (float*)(ws + WS_ACT1);

    // ---- pack all 8 GIN weight matrices to f16 WMMA fragment layout ----
    for (int l = 0; l < N_LAYERS; ++l) {
        const float* W1 = (const float*)d_in[3 + l * 8 + 0];
        const float* W2 = (const float*)d_in[3 + l * 8 + 4];
        pack_w_kernel<<<WMAT_SZ / 256, 256, 0, stream>>>(W1, Wpack + (size_t)(2 * l) * WMAT_SZ);
        pack_w_kernel<<<WMAT_SZ / 256, 256, 0, stream>>>(W2, Wpack + (size_t)(2 * l + 1) * WMAT_SZ);
    }

    (void)hipMemsetAsync(g, 0, (size_t)N_GRAPHS * N_LAYERS * FEAT * sizeof(float), stream);

    const int n4       = N_NODES * FEAT / 4;
    const int gemmGrid = (MTILES + 7) / 8;
    const float* h_cur = x;

    for (int l = 0; l < N_LAYERS; ++l) {
        const float* b1  = (const float*)d_in[3 + l * 8 + 1];
        const float* g1  = (const float*)d_in[3 + l * 8 + 2];
        const float* be1 = (const float*)d_in[3 + l * 8 + 3];
        const float* b2  = (const float*)d_in[3 + l * 8 + 5];
        const float* g2  = (const float*)d_in[3 + l * 8 + 6];
        const float* be2 = (const float*)d_in[3 + l * 8 + 7];
        const _Float16* W1p = Wpack + (size_t)(2 * l) * WMAT_SZ;
        const _Float16* W2p = Wpack + (size_t)(2 * l + 1) * WMAT_SZ;

        // z = h + sum_{edges} h[src]
        copy_f4_kernel<<<(n4 + 255) / 256, 256, 0, stream>>>((const float4*)h_cur, (float4*)z, n4);
        scatter_kernel<<<(N_EDGES * 32) / 256, 256, 0, stream>>>(h_cur, ei, z);

        // MLP part 1: pack -> WMMA GEMM -> stats
        packA_kernel<<<(MTILES * KCHUNKS * 32 + 255) / 256, 256, 0, stream>>>(z, Apack);
        gemm_wmma_kernel<<<gemmGrid, 256, 0, stream>>>(Apack, W1p, b1, y);
        (void)hipMemsetAsync(stats, 0, 2 * FEAT * sizeof(float), stream);
        bn_stats_kernel<<<(N_NODES + 255) / 256, 128, 0, stream>>>(y, stats);

        // fused BN1 + ReLU + pack, then MLP part 2
        packA_bn_kernel<<<(MTILES * KCHUNKS * 32 + 255) / 256, 256, 0, stream>>>(y, stats, g1, be1, Apack);
        gemm_wmma_kernel<<<gemmGrid, 256, 0, stream>>>(Apack, W2p, b2, y);
        (void)hipMemsetAsync(stats, 0, 2 * FEAT * sizeof(float), stream);
        bn_stats_kernel<<<(N_NODES + 255) / 256, 128, 0, stream>>>(y, stats);
        bn_relu_kernel<<<(N_NODES * FEAT + 255) / 256, 256, 0, stream>>>(y, stats, g2, be2, h);

        // pool this layer's output into g[:, l*128 : (l+1)*128]
        pool_kernel<<<(N_NODES * 32) / 256, 256, 0, stream>>>(h, batch, g, l * FEAT);

        h_cur = h;
    }

    lin1_kernel<<<(N_GRAPHS * FEAT) / 256, 256, 0, stream>>>(g, lin1_W, lin1_b, act1);
    head_kernel<<<(N_GRAPHS + 255) / 256, 256, 0, stream>>>(act1, lin2_W, lin2_b, (float*)d_out);
}
